// Attention_19859928777372
// MI455X (gfx1250) — matrix operand order repaired and checked by reading, not compiled, not run on hardware
//
#include <hip/hip_runtime.h>
#include <hip/hip_bf16.h>

// ---------------------------------------------------------------------------
// MI455X (gfx1250) attention block: bf16 WMMA + fp32 accumulate, double-
// buffered async-to-LDS staging (DMA overlapped with WMMA).
// Problem: B=4, N=2048, DIM=1024, H=16, D=64.
// ---------------------------------------------------------------------------

typedef __bf16 bf16;
typedef __attribute__((ext_vector_type(16))) __bf16 v16bf;
typedef __attribute__((ext_vector_type(8)))  float  v8f;

#define DIMC   1024
#define NHEAD  16
#define HDIM   64
#define BATCH  4
#define SEQ    2048
#define MTOT   (BATCH * SEQ)      // 8192
#define QKVN   (3 * DIMC)         // 3072

union AF { v16bf v; uint4 q[2]; };

static __device__ __forceinline__ v8f wmma_bf16(const AF& a, const AF& b, v8f c) {
    return __builtin_amdgcn_wmma_f32_16x16x32_bf16(
        false, a.v, false, b.v, (short)0, c, false, false);
}

// CDNA5 async memory->LDS copy (16B per lane), tracked by ASYNCcnt.
static __device__ __forceinline__ void async_b128(void* lds, const void* g) {
    const unsigned off = (unsigned)(uintptr_t)lds;   // low 32b of flat = LDS offset
    asm volatile("global_load_async_to_lds_b128 %0, %1, off"
                 :: "v"(off), "v"(g) : "memory");
}
static __device__ __forceinline__ void wait_async0() {
    asm volatile("s_wait_asynccnt 0x0" ::: "memory");
}

// ---------------------------------------------------------------------------
// Kernel 0a: fp32 -> bf16 (x)
// ---------------------------------------------------------------------------
__global__ void k_cvt(const float* __restrict__ s, bf16* __restrict__ d, int n) {
    int i = blockIdx.x * blockDim.x + threadIdx.x;
    const int str = gridDim.x * blockDim.x;
    for (; i < n; i += str) d[i] = (bf16)s[i];
}

// Kernel 0b: fp32 -> bf16 transpose: s[K][Nc] -> d[Nc][K]. grid=(Nc/256, K).
__global__ void k_cvt_t(const float* __restrict__ s, bf16* __restrict__ d, int Nc) {
    const int n = blockIdx.x * blockDim.x + threadIdx.x;
    const int k = blockIdx.y;
    d[(size_t)n * gridDim.y + k] = (bf16)s[(size_t)k * Nc + n];
}

// ---------------------------------------------------------------------------
// Kernel 1: QKV GEMM  x_bf16[8192,1024] @ w_qkv^T_bf16[3072,1024]
// 128x64 tile / 128-thread block (4 waves; wave = 32 rows x 64 cols, 8 WMMAs
// per k-step, B-frags reused twice). Double-buffered async staging.
// Epilogue: q,k -> [B,H,N,64]; v -> [B,H,64,N] (pre-transposed for attention).
// ---------------------------------------------------------------------------
__global__ __launch_bounds__(128)
void k_gemm_qkv(const bf16* __restrict__ X, const bf16* __restrict__ Wt,
                bf16* __restrict__ Qo, bf16* __restrict__ Ko, bf16* __restrict__ Vo) {
    __shared__ __align__(16) bf16 As[2][128 * 32];   // [m][k]
    __shared__ __align__(16) bf16 Bs[2][64 * 32];    // [n][k]

    const int t = threadIdx.x, lane = t & 31, wv = t >> 5;
    const int n0 = blockIdx.x * 64, m0 = blockIdx.y * 128;
    const int bn = t >> 1, bhf = (t & 1) * 16;

    auto stage = [&](int k0, int buf) {
        const bf16* asrc = X + (size_t)(m0 + t) * DIMC + k0;
#pragma unroll
        for (int j = 0; j < 4; ++j) async_b128(&As[buf][t * 32 + j * 8], asrc + j * 8);
        const bf16* bsrc = Wt + (size_t)(n0 + bn) * DIMC + k0 + bhf;
#pragma unroll
        for (int j = 0; j < 2; ++j)
            async_b128(&Bs[buf][bn * 32 + bhf + j * 8], bsrc + j * 8);
    };

    const v8f vz = {0.f,0.f,0.f,0.f,0.f,0.f,0.f,0.f};
    v8f acc[2][4];
#pragma unroll
    for (int i = 0; i < 2; ++i)
#pragma unroll
        for (int g = 0; g < 4; ++g) acc[i][g] = vz;

    stage(0, 0);
    const int NSTEP = DIMC / 32;
    for (int it = 0; it < NSTEP; ++it) {
        const int buf = it & 1;
        wait_async0();                    // own transfers for buf complete
        __syncthreads();                  // all waves staged buf; prev compute done
        if (it + 1 < NSTEP) stage((it + 1) * 32, buf ^ 1);   // overlap with WMMA

        AF a0, a1, b[4];
        const bf16* Ab = As[buf];
        const bf16* Bb = Bs[buf];
        const int r0 = wv * 32 + (lane & 15);
        const int kh = (lane >> 4) * 8;            // ISA A-frag K layout
        a0.q[0] = *(const uint4*)(Ab + r0 * 32 + kh);
        a0.q[1] = *(const uint4*)(Ab + r0 * 32 + 16 + kh);
        a1.q[0] = *(const uint4*)(Ab + (r0 + 16) * 32 + kh);
        a1.q[1] = *(const uint4*)(Ab + (r0 + 16) * 32 + 16 + kh);
        const int k2 = (lane >> 4) * 8;            // ISA B-frag K layout
#pragma unroll
        for (int g = 0; g < 4; ++g) {
            const int c = g * 16 + (lane & 15);
            b[g].q[0] = *(const uint4*)(Bb + c * 32 + k2);
            b[g].q[1] = *(const uint4*)(Bb + c * 32 + k2 + 16);
        }
#pragma unroll
        for (int g = 0; g < 4; ++g) {
            acc[0][g] = wmma_bf16(a0, b[g], acc[0][g]);
            acc[1][g] = wmma_bf16(a1, b[g], acc[1][g]);
        }
        __syncthreads();                  // readers done before buf is restaged
    }

    // epilogue: tile never straddles a batch boundary -> bb is block-uniform
    const int half  = lane >> 4;
    const int bb    = m0 >> 11;
    const int nbase = (m0 & (SEQ - 1)) + wv * 32;
#pragma unroll
    for (int g = 0; g < 4; ++g) {
        const int c = n0 + g * 16 + (lane & 15);
        const int which = c >> 10, cc = c & 1023;
        const int h = cc >> 6, d = cc & 63;
        const size_t hb = (size_t)(bb * NHEAD + h);
        bf16* dst = (which == 0) ? Qo : (which == 1) ? Ko : Vo;
        const size_t base = (which < 2) ? ((hb * SEQ + nbase) * HDIM + d)
                                        : ((hb * HDIM + d) * SEQ + nbase);
        const int step = (which < 2) ? HDIM : 1;
#pragma unroll
        for (int i = 0; i < 2; ++i)
#pragma unroll
            for (int r2 = 0; r2 < 8; ++r2)
                dst[base + (size_t)(i * 16 + half * 8 + r2) * step] =
                    (bf16)acc[i][g][r2];
    }
}

// ---------------------------------------------------------------------------
// Kernel 2: per-head RMSNorm + RoPE (+ q *= 1/8), in place on q/k bf16.
// ---------------------------------------------------------------------------
__global__ __launch_bounds__(256)
void k_norm_rope(bf16* __restrict__ Q, bf16* __restrict__ K,
                 const float* __restrict__ qs, const float* __restrict__ ks) {
    const int row  = (blockIdx.x * blockDim.x + threadIdx.x) >> 5;
    const int lane = threadIdx.x & 31;
    if (row >= BATCH * NHEAD * SEQ) return;
    const int n = row & (SEQ - 1);
    const size_t base = (size_t)row * HDIM;

    float q1 = (float)Q[base + lane], q2 = (float)Q[base + 32 + lane];
    float k1 = (float)K[base + lane], k2 = (float)K[base + 32 + lane];

    float sq = q1 * q1 + q2 * q2;
    float sk = k1 * k1 + k2 * k2;
#pragma unroll
    for (int m = 16; m >= 1; m >>= 1) {
        sq += __shfl_xor(sq, m, 32);
        sk += __shfl_xor(sk, m, 32);
    }
    const float rq = __frsqrt_rn(sq * (1.0f / 64.0f) + 1e-6f);
    const float rk = __frsqrt_rn(sk * (1.0f / 64.0f) + 1e-6f);
    q1 *= rq * qs[lane]; q2 *= rq * qs[lane + 32];
    k1 *= rk * ks[lane]; k2 *= rk * ks[lane + 32];

    const float invf = __powf(10000.0f, -(float)lane * (1.0f / 32.0f));
    float s, c;
    __sincosf((float)n * invf, &s, &c);

    Q[base + lane]      = (bf16)((q1 * c - q2 * s) * 0.125f);
    Q[base + 32 + lane] = (bf16)((q2 * c + q1 * s) * 0.125f);
    K[base + lane]      = (bf16)(k1 * c - k2 * s);
    K[base + 32 + lane] = (bf16)(k2 * c + k1 * s);
}

// ---------------------------------------------------------------------------
// Kernel 3: flash attention. WG = 4 waves / 128 thr per (bh, 64-query block).
// Q frags register-resident; K [key][d] and V^T [d][key] double-buffered via
// async DMA (V pre-transposed globally -> staging is a straight copy).
// ---------------------------------------------------------------------------
__global__ __launch_bounds__(128)
void k_attn(const bf16* __restrict__ Q, const bf16* __restrict__ K,
            const bf16* __restrict__ Vt_g, bf16* __restrict__ O) {
    __shared__ __align__(16) bf16  Kt[2][64 * 64];    // [key][d]
    __shared__ __align__(16) bf16  Vt[2][64 * 64];    // [d][key]
    __shared__ __align__(16) float Sb[4][16 * 64];
    __shared__ __align__(16) bf16  Pb[4][16 * 64];
    __shared__ float rstat[4][16];

    const int t = threadIdx.x, lane = t & 31, wv = t >> 5;
    const int bh = blockIdx.y;
    const int qrow0 = blockIdx.x * 64 + wv * 16;
    const size_t hbase = (size_t)bh * SEQ * HDIM;
    const int sr = t >> 1, shf = (t & 1) * 32;        // staging role

    auto stageKV = [&](int kb, int buf) {
        const bf16* kp = K    + hbase + (size_t)(kb + sr) * HDIM + shf;
        const bf16* vp = Vt_g + hbase + (size_t)sr * SEQ + kb + shf;
#pragma unroll
        for (int j = 0; j < 4; ++j) {
            async_b128(&Kt[buf][sr * 64 + shf + j * 8], kp + j * 8);
            async_b128(&Vt[buf][sr * 64 + shf + j * 8], vp + j * 8);
        }
    };

    AF aQ0, aQ1;
    {
        const int r  = qrow0 + (lane & 15);
        const int kh = (lane >> 4) * 8;
        const bf16* qp = Q + hbase + (size_t)r * HDIM;
        aQ0.q[0] = *(const uint4*)(qp + kh);
        aQ0.q[1] = *(const uint4*)(qp + 16 + kh);
        aQ1.q[0] = *(const uint4*)(qp + 32 + kh);
        aQ1.q[1] = *(const uint4*)(qp + 48 + kh);
    }

    const v8f vz = {0.f,0.f,0.f,0.f,0.f,0.f,0.f,0.f};
    v8f o[4];
#pragma unroll
    for (int g = 0; g < 4; ++g) o[g] = vz;
    float m_run = -3.0e38f, l_run = 0.f;
    const int srow = lane & 15, shalf = lane >> 4;

    stageKV(0, 0);
    const int NBLK = SEQ / 64;
    for (int it = 0; it < NBLK; ++it) {
        const int buf = it & 1;
        wait_async0();
        __syncthreads();                  // buf staged everywhere; compute(it-1) done
        if (it + 1 < NBLK) stageKV((it + 1) * 64, buf ^ 1);

        // S = Q K^T (16q x 64k), dump to LDS for softmax
#pragma unroll
        for (int g = 0; g < 4; ++g) {
            v8f s = vz;
            AF b0, b1;
            const bf16* Kb = Kt[buf];
            const int keyl = g * 16 + (lane & 15);
            const int k2   = (lane >> 4) * 8;
            b0.q[0] = *(const uint4*)(Kb + keyl * 64 + k2);
            b0.q[1] = *(const uint4*)(Kb + keyl * 64 + k2 + 16);
            b1.q[0] = *(const uint4*)(Kb + keyl * 64 + 32 + k2);
            b1.q[1] = *(const uint4*)(Kb + keyl * 64 + 32 + k2 + 16);
            s = wmma_bf16(aQ0, b0, s);
            s = wmma_bf16(aQ1, b1, s);
#pragma unroll
            for (int r2 = 0; r2 < 8; ++r2)
                Sb[wv][(shalf * 8 + r2) * 64 + g * 16 + (lane & 15)] = s[r2];
        }
        __syncthreads();

        { // online softmax: lane owns row=srow, 32 of its 64 columns
            const float* sp = &Sb[wv][srow * 64 + shalf * 32];
            float mx = -3.0e38f;
#pragma unroll
            for (int j = 0; j < 32; ++j) mx = fmaxf(mx, sp[j]);
            mx = fmaxf(mx, __shfl_xor(mx, 16, 32));
            const float m_new = fmaxf(m_run, mx);
            float sum = 0.f;
            bf16* pp = &Pb[wv][srow * 64 + shalf * 32];
#pragma unroll
            for (int j = 0; j < 32; ++j) {
                const float p = __expf(sp[j] - m_new);
                sum += p;
                pp[j] = (bf16)p;
            }
            sum += __shfl_xor(sum, 16, 32);
            const float alpha = __expf(m_run - m_new);
            l_run = l_run * alpha + sum;
            m_run = m_new;
            if (lane < 16) rstat[wv][srow] = alpha;
        }
        __syncthreads();

#pragma unroll
        for (int r2 = 0; r2 < 8; ++r2) {
            const float a = rstat[wv][shalf * 8 + r2];
#pragma unroll
            for (int g = 0; g < 4; ++g) o[g][r2] *= a;
        }

        // O += P V  (2 K-chunks x 4 d-tiles = 8 WMMAs)
#pragma unroll
        for (int kk = 0; kk < 64; kk += 32) {
            AF ap;
            const int kh = (lane >> 4) * 8;
            const bf16* pr = &Pb[wv][(lane & 15) * 64 + kk];
            ap.q[0] = *(const uint4*)(pr + kh);
            ap.q[1] = *(const uint4*)(pr + 16 + kh);
#pragma unroll
            for (int g = 0; g < 4; ++g) {
                AF bv;
                const bf16* Vb = Vt[buf];
                const int d  = g * 16 + (lane & 15);
                const int k2 = (lane >> 4) * 8 + kk;
                bv.q[0] = *(const uint4*)(Vb + d * 64 + k2);
                bv.q[1] = *(const uint4*)(Vb + d * 64 + k2 + 16);
                o[g] = wmma_bf16(ap, bv, o[g]);
            }
        }
        __syncthreads();                  // PV readers done before buf restage
    }

    if (lane < 16) rstat[wv][srow] = 1.0f / l_run;
    __syncthreads();
    const int b = bh >> 4, h = bh & 15;
#pragma unroll
    for (int r2 = 0; r2 < 8; ++r2) {
        const float inv = rstat[wv][shalf * 8 + r2];
        const int n = qrow0 + shalf * 8 + r2;
#pragma unroll
        for (int g = 0; g < 4; ++g) {
            const int d = g * 16 + (lane & 15);
            O[((size_t)(b * SEQ + n) * DIMC) + h * HDIM + d] = (bf16)(o[g][r2] * inv);
        }
    }
}

// ---------------------------------------------------------------------------
// Kernel 4: output projection attn[8192,1024] @ w_proj^T[1024,1024] + bias.
// ---------------------------------------------------------------------------
__global__ __launch_bounds__(128)
void k_gemm_proj(const bf16* __restrict__ X, const bf16* __restrict__ Wt,
                 const float* __restrict__ bias, float* __restrict__ out) {
    __shared__ __align__(16) bf16 As[2][128 * 32];
    __shared__ __align__(16) bf16 Bs[2][64 * 32];

    const int t = threadIdx.x, lane = t & 31, wv = t >> 5;
    const int n0 = blockIdx.x * 64, m0 = blockIdx.y * 128;
    const int bn = t >> 1, bhf = (t & 1) * 16;

    auto stage = [&](int k0, int buf) {
        const bf16* asrc = X + (size_t)(m0 + t) * DIMC + k0;
#pragma unroll
        for (int j = 0; j < 4; ++j) async_b128(&As[buf][t * 32 + j * 8], asrc + j * 8);
        const bf16* bsrc = Wt + (size_t)(n0 + bn) * DIMC + k0 + bhf;
#pragma unroll
        for (int j = 0; j < 2; ++j)
            async_b128(&Bs[buf][bn * 32 + bhf + j * 8], bsrc + j * 8);
    };

    const v8f vz = {0.f,0.f,0.f,0.f,0.f,0.f,0.f,0.f};
    v8f acc[2][4];
#pragma unroll
    for (int i = 0; i < 2; ++i)
#pragma unroll
        for (int g = 0; g < 4; ++g) acc[i][g] = vz;

    stage(0, 0);
    const int NSTEP = DIMC / 32;
    for (int it = 0; it < NSTEP; ++it) {
        const int buf = it & 1;
        wait_async0();
        __syncthreads();
        if (it + 1 < NSTEP) stage((it + 1) * 32, buf ^ 1);

        AF a0, a1, b[4];
        const bf16* Ab = As[buf];
        const bf16* Bb = Bs[buf];
        const int r0 = wv * 32 + (lane & 15);
        const int kh = (lane >> 4) * 8;
        a0.q[0] = *(const uint4*)(Ab + r0 * 32 + kh);
        a0.q[1] = *(const uint4*)(Ab + r0 * 32 + 16 + kh);
        a1.q[0] = *(const uint4*)(Ab + (r0 + 16) * 32 + kh);
        a1.q[1] = *(const uint4*)(Ab + (r0 + 16) * 32 + 16 + kh);
        const int k2 = (lane >> 4) * 8;
#pragma unroll
        for (int g = 0; g < 4; ++g) {
            const int c = g * 16 + (lane & 15);
            b[g].q[0] = *(const uint4*)(Bb + c * 32 + k2);
            b[g].q[1] = *(const uint4*)(Bb + c * 32 + k2 + 16);
        }
#pragma unroll
        for (int g = 0; g < 4; ++g) {
            acc[0][g] = wmma_bf16(a0, b[g], acc[0][g]);
            acc[1][g] = wmma_bf16(a1, b[g], acc[1][g]);
        }
        __syncthreads();
    }

    const int half = lane >> 4;
#pragma unroll
    for (int g = 0; g < 4; ++g) {
        const int c = n0 + g * 16 + (lane & 15);
        const float bv = bias[c];
#pragma unroll
        for (int i = 0; i < 2; ++i)
#pragma unroll
            for (int r2 = 0; r2 < 8; ++r2) {
                const int m = m0 + wv * 32 + i * 16 + half * 8 + r2;
                out[(size_t)m * DIMC + c] = acc[i][g][r2] + bv;
            }
    }
}

// ---------------------------------------------------------------------------
// Orchestration (all on `stream`, graph-capture safe). Workspace ~92 MB:
//   x_bf16[8192,1024] @0 | wqkvT[3072,1024] @16M | wprojT[1024,1024] @22M
//   q,k [B,H,N,64] @24M/@40M | vT [B,H,64,N] @56M | attn[8192,1024] @72M
// ---------------------------------------------------------------------------
extern "C" void kernel_launch(void* const* d_in, const int* in_sizes, int n_in,
                              void* d_out, int out_size, void* d_ws, size_t ws_size,
                              hipStream_t stream) {
    (void)in_sizes; (void)n_in; (void)out_size; (void)ws_size;
    const float* x       = (const float*)d_in[0];
    const float* w_qkv   = (const float*)d_in[1];
    const float* q_scale = (const float*)d_in[2];
    const float* k_scale = (const float*)d_in[3];
    const float* w_proj  = (const float*)d_in[4];
    const float* b_proj  = (const float*)d_in[5];
    float* out = (float*)d_out;

    char* w = (char*)d_ws;
    bf16* xb  = (bf16*)(w);
    bf16* wqb = (bf16*)(w + (size_t)16777216);
    bf16* wpb = (bf16*)(w + (size_t)23068672);
    bf16* qb  = (bf16*)(w + (size_t)25165824);
    bf16* kb  = (bf16*)(w + (size_t)41943040);
    bf16* vtb = (bf16*)(w + (size_t)58720256);
    bf16* ab  = (bf16*)(w + (size_t)75497472);

    k_cvt  <<<1024, 256, 0, stream>>>(x, xb, MTOT * DIMC);
    k_cvt_t<<<dim3(QKVN / 256, DIMC), 256, 0, stream>>>(w_qkv,  wqb, QKVN);
    k_cvt_t<<<dim3(DIMC / 256, DIMC), 256, 0, stream>>>(w_proj, wpb, DIMC);

    k_gemm_qkv <<<dim3(QKVN / 64, MTOT / 128), 128, 0, stream>>>(xb, wqb, qb, kb, vtb);
    k_norm_rope<<<(BATCH * NHEAD * SEQ * 32) / 256, 256, 0, stream>>>(qb, kb, q_scale, k_scale);
    k_attn     <<<dim3(SEQ / 64, BATCH * NHEAD), 128, 0, stream>>>(qb, kb, vtb, ab);
    k_gemm_proj<<<dim3(DIMC / 64, MTOT / 128), 128, 0, stream>>>(ab, wpb, b_proj, out);
}
